// EquiPocket_56693568307780
// MI455X (gfx1250) — compile-verified, matrix operand order checked by reading, not run on hardware
//
#include <hip/hip_runtime.h>

typedef __attribute__((ext_vector_type(16))) _Float16 v16h;
typedef __attribute__((ext_vector_type(8)))  _Float16 h8;
typedef __attribute__((ext_vector_type(8)))  float    v8f;
typedef __attribute__((ext_vector_type(4)))  float    f4;

#define ACT_NONE 0
#define ACT_RELU 1
#define ACT_SILU 2
#define ACT_SIG  3

struct GSeg  { const float* ptr; const int* idx; int width; int ld; };
struct GemmArgs {
  GSeg seg[4];
  int  nseg, K, M, N, ldw, ldo, act;
  const _Float16* W;     // transposed+padded: [Npad32][Kpad32], row n = col n of W[K,N]
  const float*    bias;  // [N] or null
  const float*    resid; // [M,ldo] or null
  float*          out;   // [M,ldo]
};

static inline int ceildiv(int a, int b) { return (a + b - 1) / b; }
static inline GSeg S(const float* p, int w, int ld, const int* idx = nullptr) {
  GSeg s; s.ptr = p; s.idx = idx; s.width = w; s.ld = ld; return s;
}

__device__ __forceinline__ float act_apply(float x, int act) {
  if (act == ACT_RELU) return x > 0.f ? x : 0.f;
  if (act == ACT_SILU) return x / (1.f + __expf(-x));
  if (act == ACT_SIG)  return 1.f / (1.f + __expf(-x));
  return x;
}

// per-lane A-row segment bases (resolved gather), cumulative widths c0<=c1<=c2<=K
struct ABases {
  const float* b0; const float* b1; const float* b2; const float* b3;
  int c0, c1, c2, K;
};

// Main-loop A fragment: two runs of 8 consecutive k per lane.
// Requirements (hold for every layer in this pipeline):
//  - every cumulative boundary < Kmain is a multiple of 8 (128/256 here; the odd
//    boundaries 257/258 and narrow segments only appear in the tail step)
//  - every segment addressed in the main loop has 16B-aligned rows
//    (ld in {64,128,640} on 256B-aligned buffers) -> aligned float4 loads
template <bool MULTI>
__device__ __forceinline__ v16h a_frag_main(const ABases& A, int k0, int hi) {
  const float* q0; const float* q1;
#pragma unroll
  for (int gi = 0; gi < 2; ++gi) {
    const int kr = k0 + 8 * hi + 16 * gi;
    const float* p = A.b0;
    int sub = 0;
    if (MULTI) {
      p   = (kr >= A.c0) ? A.b1 : p;
      sub = (kr >= A.c0) ? A.c0 : sub;
      p   = (kr >= A.c1) ? A.b2 : p;
      sub = (kr >= A.c1) ? A.c1 : sub;
      p   = (kr >= A.c2) ? A.b3 : p;
      sub = (kr >= A.c2) ? A.c2 : sub;
    }
    const float* q = p + (kr - sub);
    if (gi == 0) q0 = q; else q1 = q;
  }
  // issue all four 16B loads before any conversion (clause-friendly)
  f4 t0 = *(const f4*)(q0);
  f4 t1 = *(const f4*)(q0 + 4);
  f4 t2 = *(const f4*)(q1);
  f4 t3 = *(const f4*)(q1 + 4);
  v16h a;
#pragma unroll
  for (int j = 0; j < 4; ++j) {
    a[j]      = (_Float16)t0[j];
    a[4 + j]  = (_Float16)t1[j];
    a[8 + j]  = (_Float16)t2[j];
    a[12 + j] = (_Float16)t3[j];
  }
  return a;
}

// Tail A fragment (last partial K-step): fully guarded scalar loads, pad slots
// forced to exact 0 (padded B (=0) must never meet Inf/NaN: WMMA 0*Inf -> NaN).
__device__ __forceinline__ v16h a_frag_tail(const ABases& A, int k0, int hi) {
  v16h a;
#pragma unroll
  for (int e = 0; e < 16; ++e) {
    const int k = k0 + (e & 7) + 8 * hi + 16 * (e >> 3);
    const float* p = A.b0;
    int sub = 0;
    p   = (k >= A.c0) ? A.b1 : p;
    sub = (k >= A.c0) ? A.c0 : sub;
    p   = (k >= A.c1) ? A.b2 : p;
    sub = (k >= A.c1) ? A.c1 : sub;
    p   = (k >= A.c2) ? A.b3 : p;
    sub = (k >= A.c2) ? A.c2 : sub;
    const bool pad = k >= A.K;
    const int o = pad ? 0 : (k - sub);
    const float* q = pad ? A.b0 : p;
    float av = q[o];
    av = pad ? 0.f : av;
    a[e] = (_Float16)av;
  }
  return a;
}

// ---------------------------------------------------------------------------
// Generic WMMA GEMM: out[M,N] = act(concat_segments(A)[M,K] @ W[K,N] + b (+resid))
// 4 waves/block; each wave owns a 16x32 output tile (2 WMMAs per K-step of 32).
// ---------------------------------------------------------------------------
__global__ void __launch_bounds__(128) wmma_gemm(GemmArgs g) {
  const int wave = threadIdx.x >> 5;
  const int lane = threadIdx.x & 31;
  const int row0 = (blockIdx.x * 4 + wave) * 16;
  const int col0 = blockIdx.y * 32;
  if (row0 >= g.M) return;               // uniform per wave: EXEC stays all-ones
  const int lo = lane & 15, hi = lane >> 4;
  int arow = row0 + lo; if (arow >= g.M) arow = g.M - 1;   // garbage rows are store-masked
  const int bcol0 = col0 + lo;
  const int bcol1 = col0 + 16 + lo;

  ABases A;
  { const GSeg& s = g.seg[0]; int r = s.idx ? s.idx[arow] : arow;
    A.b0 = s.ptr + (long)r * s.ld; A.c0 = s.width; }
  A.b1 = A.b0; A.b2 = A.b0; A.b3 = A.b0; A.c1 = A.c0; A.c2 = A.c0; A.K = g.K;
  if (g.nseg > 1) { const GSeg& s = g.seg[1]; int r = s.idx ? s.idx[arow] : arow;
    A.b1 = s.ptr + (long)r * s.ld; A.c1 = A.c0 + s.width; A.c2 = A.c1; }
  if (g.nseg > 2) { const GSeg& s = g.seg[2]; int r = s.idx ? s.idx[arow] : arow;
    A.b2 = s.ptr + (long)r * s.ld; A.c2 = A.c1 + s.width; }
  if (g.nseg > 3) { const GSeg& s = g.seg[3]; int r = s.idx ? s.idx[arow] : arow;
    A.b3 = s.ptr + (long)r * s.ld; }

  const _Float16* w0p = g.W + (long)bcol0 * g.ldw;
  const _Float16* w1p = g.W + (long)bcol1 * g.ldw;

  v8f acc0, acc1;
  const float bv0 = (g.bias && bcol0 < g.N) ? g.bias[bcol0] : 0.f;
  const float bv1 = (g.bias && bcol1 < g.N) ? g.bias[bcol1] : 0.f;
#pragma unroll
  for (int v = 0; v < 8; ++v) { acc0[v] = bv0; acc1[v] = bv1; }

  auto step = [&](v16h a, int k0) {
    const int kb = k0 + 8 * hi;
    // group all four 16B B loads before the WMMAs
    h8 x0 = *(const h8*)(w0p + kb);
    h8 x1 = *(const h8*)(w0p + kb + 16);
    h8 y0 = *(const h8*)(w1p + kb);
    h8 y1 = *(const h8*)(w1p + kb + 16);
    v16h bf0 = __builtin_shufflevector(x0, x1, 0,1,2,3,4,5,6,7,8,9,10,11,12,13,14,15);
    v16h bf1 = __builtin_shufflevector(y0, y1, 0,1,2,3,4,5,6,7,8,9,10,11,12,13,14,15);
    acc0 = __builtin_amdgcn_wmma_f32_16x16x32_f16(false, a, false, bf0, (short)0, acc0, false, false);
    acc1 = __builtin_amdgcn_wmma_f32_16x16x32_f16(false, a, false, bf1, (short)0, acc1, false, false);
  };

  const int Kmain = g.K & ~31;
  if (g.nseg == 1) {               // scalar-uniform branch
    for (int k0 = 0; k0 < Kmain; k0 += 32) step(a_frag_main<false>(A, k0, hi), k0);
  } else {
    for (int k0 = 0; k0 < Kmain; k0 += 32) step(a_frag_main<true>(A, k0, hi), k0);
  }
  if (Kmain < g.K) step(a_frag_tail(A, Kmain, hi), Kmain);

#pragma unroll
  for (int v = 0; v < 8; ++v) {
    const int m = row0 + v + 8 * hi;
    if (m < g.M) {
      const int n0 = col0 + lo;
      if (n0 < g.N) {
        float xv = acc0[v];
        if (g.resid) xv += g.resid[(long)m * g.ldo + n0];
        g.out[(long)m * g.ldo + n0] = act_apply(xv, g.act);
      }
      const int n1 = col0 + 16 + lo;
      if (n1 < g.N) {
        float xv = acc1[v];
        if (g.resid) xv += g.resid[(long)m * g.ldo + n1];
        g.out[(long)m * g.ldo + n1] = act_apply(xv, g.act);
      }
    }
  }
}

static void run_gemm(hipStream_t st, const GSeg* segs, int nseg,
                     const _Float16* W, const float* bias, const float* resid,
                     float* out, int ldo, int M, int Nn, int act) {
  GemmArgs g;
  int K = 0;
  for (int i = 0; i < nseg; ++i) { g.seg[i] = segs[i]; K += segs[i].width; }
  for (int i = nseg; i < 4; ++i) { g.seg[i] = GSeg{nullptr, nullptr, 0, 0}; }
  g.nseg = nseg; g.K = K; g.M = M; g.N = Nn; g.ldo = ldo; g.act = act;
  g.ldw = (K + 31) & ~31;
  g.W = W; g.bias = bias; g.resid = resid; g.out = out;
  dim3 grid(ceildiv(M, 64), ceildiv(Nn, 32));
  wmma_gemm<<<grid, dim3(128), 0, st>>>(g);
}

// ---------------------------------------------------------------------------
// Weight prep: W[K,N] f32 -> Wt[Npad32][Kpad32] f16 (transposed, zero padded)
// ---------------------------------------------------------------------------
__global__ void k_cvt_wt(const float* src, _Float16* dst, int K, int N,
                         int Kpad, int Npad) {
  long i = (long)blockIdx.x * 256 + threadIdx.x;
  if (i >= (long)Npad * Kpad) return;
  int n = (int)(i / Kpad), k = (int)(i % Kpad);
  float v = (n < N && k < K) ? src[(long)k * N + n] : 0.f;
  dst[i] = (_Float16)v;
}

// ---------------------------------------------------------------------------
// Elementwise / graph kernels
// ---------------------------------------------------------------------------
__global__ void k_fill_u32(unsigned* p, unsigned v, int n) {
  int i = blockIdx.x * 256 + threadIdx.x;
  if (i < n) p[i] = v;
}

__global__ void k_relu(float* p, long n) {
  long i = (long)blockIdx.x * 256 + threadIdx.x;
  if (i < n) { float v = p[i]; p[i] = v > 0.f ? v : 0.f; }
}

// mean+max pooling over Vp contiguous vertices per surface atom
__global__ void k_pool_verts(const float* lg, const float* sd, float* geom,
                             float* ssz, int Ns, int Vp) {
  int s = blockIdx.x, f = threadIdx.x;
  if (s >= Ns) return;
  if (f < 64) {
    float sum = 0.f, mx = -3.4e38f;
    for (int v = 0; v < Vp; ++v) {
      float val = lg[((long)s * Vp + v) * 64 + f];
      sum += val; mx = val > mx ? val : mx;
    }
    geom[(long)s * 128 + f] = sum / (float)Vp;
    geom[(long)s * 128 + 64 + f] = mx;
  }
  if (f < 7) {
    float sum = 0.f, mx = -3.4e38f;
    for (int v = 0; v < Vp; ++v) {
      float val = sd[((long)s * Vp + v) * 7 + f];
      sum += val; mx = val > mx ? val : mx;
    }
    ssz[(long)s * 14 + f] = sum / (float)Vp;
    ssz[(long)s * 14 + 7 + f] = mx;
  }
}

__global__ void k_gat_dots(const float* hg, const float* as, const float* ad,
                           float* asd, float* add_, int N) {
  int n = blockIdx.x * 256 + threadIdx.x;
  if (n >= N) return;
  float a = 0.f, b = 0.f;
  for (int t = 0; t < 128; ++t) {
    float h = hg[(long)n * 128 + t];
    a += h * as[t]; b += h * ad[t];
  }
  asd[n] = a; add_[n] = b;
}

__device__ __forceinline__ unsigned enc_f(float v) {
  unsigned b = __float_as_uint(v);
  return (b & 0x80000000u) ? ~b : (b | 0x80000000u);
}
__device__ __forceinline__ float dec_f(unsigned u) {
  unsigned b = (u & 0x80000000u) ? (u & 0x7FFFFFFFu) : ~u;
  return __uint_as_float(b);
}

__global__ void k_gat_logit(const float* asd, const float* add_, const int* src,
                            const int* dst, float* logit, unsigned* mx, int E) {
  int e = blockIdx.x * 256 + threadIdx.x;
  if (e >= E) return;
  float l = asd[src[e]] + add_[dst[e]];
  l = l > 0.f ? l : 0.2f * l;               // leaky_relu(., 0.2)
  logit[e] = l;
  atomicMax(&mx[dst[e]], enc_f(l));
}

__global__ void k_gat_exp(float* logit, const unsigned* mx, const int* dst,
                          float* denom, int E) {
  int e = blockIdx.x * 256 + threadIdx.x;
  if (e >= E) return;
  float v = __expf(logit[e] - dec_f(mx[dst[e]]));
  logit[e] = v;
  atomicAdd(&denom[dst[e]], v);
}

__global__ void k_gat_agg(const float* ew, const float* denom, const float* hg,
                          const int* src, const int* dst, float* hout, int E) {
  int e = blockIdx.x, t = threadIdx.x;
  if (e >= E) return;
  int d = dst[e];
  float w = ew[e] / (denom[d] + 1e-16f);
  atomicAdd(&hout[(long)d * 128 + t], w * hg[(long)src[e] * 128 + t]);
}

__global__ void k_edge_geom_bond(const float* cpos, const int* src, const int* dst,
                                 float* diff, float* d2, int E) {
  int e = blockIdx.x * 256 + threadIdx.x;
  if (e >= E) return;
  int s = src[e], d = dst[e];
  float acc = 0.f;
  for (int c = 0; c < 3; ++c) {
    float dv = cpos[(long)s * 3 + c] - cpos[(long)d * 3 + c];
    diff[(long)e * 3 + c] = dv;
    acc += dv * dv;
  }
  d2[e] = acc;
}

__global__ void k_edge_geom_surf(const float* xc, const int* src, const int* dst,
                                 float* diff, float* r2, int E) {
  int e = blockIdx.x * 256 + threadIdx.x;
  if (e >= E) return;
  int s = src[e], d = dst[e];
  for (int ch = 0; ch < 2; ++ch) {
    float acc = 0.f;
    for (int c = 0; c < 3; ++c) {
      float dv = xc[(long)s * 6 + ch * 3 + c] - xc[(long)d * 6 + ch * 3 + c];
      diff[(long)e * 6 + ch * 3 + c] = dv;
      acc += dv * dv;
    }
    r2[(long)e * 2 + ch] = acc;
  }
}

// acc[dst, comp] += diff[e, comp] * coef[e, comp/3];   cnt[dst] += 1
__global__ void k_coord_acc(const float* diff, const float* coef, const int* dst,
                            float* acc, float* cnt, int E, int C) {
  int D = 3 * C;
  long i = (long)blockIdx.x * 256 + threadIdx.x;
  if (i >= (long)E * D) return;
  int e = (int)(i / D), comp = (int)(i % D), ch = comp / 3;
  int d = dst[e];
  atomicAdd(&acc[(long)d * D + comp], diff[(long)e * D + comp] * coef[(long)e * C + ch]);
  if (comp == 0) atomicAdd(&cnt[d], 1.f);
}

__global__ void k_apply_coord(float* pos, const float* acc, const float* cnt,
                              int N, int D) {
  long i = (long)blockIdx.x * 256 + threadIdx.x;
  if (i >= (long)N * D) return;
  int n = (int)(i / D);
  float c = cnt[n]; c = c < 1.f ? 1.f : c;
  pos[i] += acc[i] / c;
}

__global__ void k_seg_sum128(const float* m, const int* dst, float* agg, int E) {
  int e = blockIdx.x, t = threadIdx.x;
  if (e >= E) return;
  atomicAdd(&agg[(long)dst[e] * 128 + t], m[(long)e * 128 + t]);
}

__global__ void k_init_xc(const float* pos, const float* scp, const int* sidx,
                          float* xc, int Ns) {
  long i = (long)blockIdx.x * 256 + threadIdx.x;
  if (i >= (long)Ns * 3) return;
  int s = (int)(i / 3), c = (int)(i % 3);
  xc[(long)s * 6 + c]     = pos[(long)sidx[s] * 3 + c];
  xc[(long)s * 6 + 3 + c] = scp[(long)s * 3 + c];
}

__global__ void k_gate(float* hs, const float* atts, int Ns) {
  int s = blockIdx.x, t = threadIdx.x;
  if (s >= Ns) return;
  for (int b = 0; b < 5; ++b)
    hs[(long)s * 640 + b * 128 + t] *= atts[(long)s * 5 + b];
}

__global__ void k_angle(const float* xc, const float* pos, const int* sidx,
                        float* out, int Ns) {
  long i = (long)blockIdx.x * 256 + threadIdx.x;
  if (i >= (long)Ns * 3) return;
  int s = (int)(i / 3), c = (int)(i % 3);
  out[i] = xc[(long)s * 6 + c] - pos[(long)sidx[s] * 3 + c];
}

// ---------------------------------------------------------------------------
// Host orchestration
// ---------------------------------------------------------------------------
extern "C" void kernel_launch(void* const* d_in, const int* in_sizes, int n_in,
                              void* d_out, int out_size, void* d_ws, size_t ws_size,
                              hipStream_t stream) {
  (void)n_in; (void)out_size; (void)ws_size;
  auto Pf = [&](int i) { return (const float*)d_in[i]; };

  // tensors after the 112 flattened param leaves (setup_inputs insertion order)
  const float* x     = Pf(112);
  const float* pos   = Pf(113);
  const float* eattr = Pf(114);
  const float* sd    = Pf(115);
  const float* cr    = Pf(116);
  const float* scp   = Pf(117);
  const int* bei  = (const int*)d_in[118];
  const int* sei  = (const int*)d_in[119];
  const int* sidx = (const int*)d_in[121];

  const int N  = in_sizes[112] / 6;
  const int Eb = in_sizes[114] / 3;
  const int Nv = in_sizes[115] / 7;
  const int Ns = in_sizes[117] / 3;
  const int Es = in_sizes[119] / 2;
  const int Vp = Nv / Ns;
  const int* bsrc = bei;  const int* bdst = bei + Eb;
  const int* ssrc = sei;  const int* sdst = sei + Es;

  unsigned char* wsp = (unsigned char*)d_ws;
  auto alloc = [&](size_t bytes) -> void* {
    void* p = (void*)wsp; wsp += (bytes + 255) & ~(size_t)255; return p;
  };
  auto af = [&](size_t elems) { return (float*)alloc(elems * 4); };
  // transpose + pad + f16-convert one weight matrix [K,N] -> [Npad32][Kpad32]
  auto cvt = [&](int pi, int Kd, int Nd) -> const _Float16* {
    int Kpad = (Kd + 31) & ~31, Npad = (Nd + 31) & ~31;
    long tot = (long)Npad * Kpad;
    _Float16* d = (_Float16*)alloc((size_t)tot * 2);
    k_cvt_wt<<<dim3((int)((tot + 255) / 256)), dim3(256), 0, stream>>>(
        Pf(pi), d, Kd, Nd, Kpad, Npad);
    return d;
  };

  // ---- weights -> f16 (param leaf layout: insertion order, W then b)
  const _Float16* w_lg1 = cvt(0, 7, 64);      const float* b_lg1 = Pf(1);
  const _Float16* w_lg2 = cvt(2, 64, 64);     const float* b_lg2 = Pf(3);
  const _Float16* w_sf1 = cvt(4, 14, 128);    const float* b_sf1 = Pf(5);
  const _Float16* w_sf2 = cvt(6, 128, 128);   const float* b_sf2 = Pf(7);
  const _Float16* w_ge1 = cvt(8, 384, 128);   const float* b_ge1 = Pf(9);
  const _Float16* w_ge2 = cvt(10, 128, 128);  const float* b_ge2 = Pf(11);
  const _Float16* w_at1 = cvt(12, 6, 128);    const float* b_at1 = Pf(13);
  const _Float16* w_at2 = cvt(14, 128, 5);    const float* b_at2 = Pf(15);
  const _Float16* w_ou1 = cvt(16, 640, 640);  const float* b_ou1 = Pf(17);
  const _Float16* w_ou2 = cvt(18, 640, 1);    const float* b_ou2 = Pf(19);
  const _Float16* w_atm = cvt(20, 6, 16);     const float* b_atm = Pf(21);
  const _Float16* w_bnd = cvt(22, 3, 16);     const float* b_bnd = Pf(23);
  const _Float16* w_gat = cvt(24, 16, 128);   const float* b_gat = Pf(25);
  const float* gat_as = Pf(26);               const float* gat_ad = Pf(27);

  const _Float16 *w_me1[3], *w_me2[3], *w_mx1[3], *w_mx2[3], *w_mh1[3], *w_mh2[3];
  const float    *b_me1[3], *b_me2[3], *b_mx1[3], *b_mx2[3], *b_mh1[3], *b_mh2[3];
  for (int l = 0; l < 3; ++l) {
    int p = 28 + 12 * l;
    w_me1[l] = cvt(p + 0, 273, 128);  b_me1[l] = Pf(p + 1);
    w_me2[l] = cvt(p + 2, 128, 128);  b_me2[l] = Pf(p + 3);
    w_mx1[l] = cvt(p + 4, 128, 128);  b_mx1[l] = Pf(p + 5);
    w_mx2[l] = cvt(p + 6, 128, 1);    b_mx2[l] = Pf(p + 7);
    w_mh1[l] = cvt(p + 8, 256, 128);  b_mh1[l] = Pf(p + 9);
    w_mh2[l] = cvt(p + 10, 128, 128); b_mh2[l] = Pf(p + 11);
  }
  const _Float16 *w_se1[4], *w_se2[4], *w_sx1[4], *w_sx2[4], *w_sh1[4], *w_sh2[4];
  const float    *b_se1[4], *b_se2[4], *b_sx1[4], *b_sx2[4], *b_sh1[4], *b_sh2[4];
  for (int l = 0; l < 4; ++l) {
    int p = 64 + 12 * l;
    w_se1[l] = cvt(p + 0, 258, 128);  b_se1[l] = Pf(p + 1);
    w_se2[l] = cvt(p + 2, 128, 128);  b_se2[l] = Pf(p + 3);
    w_sx1[l] = cvt(p + 4, 128, 128);  b_sx1[l] = Pf(p + 5);
    w_sx2[l] = cvt(p + 6, 128, 2);    b_sx2[l] = Pf(p + 7);
    w_sh1[l] = cvt(p + 8, 256, 128);  b_sh1[l] = Pf(p + 9);
    w_sh2[l] = cvt(p + 10, 128, 128); b_sh2[l] = Pf(p + 11);
  }

  // ---- activation scratch
  size_t Emax = (size_t)(Es > Eb ? Es : Eb);
  size_t big = Emax * 128;
  if ((size_t)Nv * 64 > big) big = (size_t)Nv * 64;
  if ((size_t)Ns * 640 > big) big = (size_t)Ns * 640;
  float* bufA  = af(big);
  float* bufB  = af(big);
  float* geom  = af((size_t)Ns * 128);
  float* ssz   = af((size_t)Ns * 14);
  float* sse   = af((size_t)Ns * 128);
  float* h0    = af((size_t)N * 16);
  float* ebb   = af((size_t)Eb * 16);
  float* hg    = af((size_t)N * 128);
  float* asd   = af(N);
  float* addt  = af(N);
  unsigned* mx = (unsigned*)alloc((size_t)N * 4);
  float* denom = af(N);
  float* logit = af(Eb);
  float* hA    = af((size_t)N * 128);
  float* hB    = af((size_t)N * 128);
  float* cpos  = af((size_t)N * 3);
  float* diffb = af(Emax * 6);
  float* r2b   = af(Emax * 2);
  float* coef  = af(Emax * 2);
  float* cacc  = af((size_t)N * 6);
  float* ccnt  = af(N);
  float* aggN  = af((size_t)N * 128);
  float* u1    = af((size_t)N * 128);
  float* hsbuf = af((size_t)Ns * 640);
  float* xc    = af((size_t)Ns * 6);
  float* atts  = af((size_t)Ns * 5);

  // ==== 1. local geometric modeling: vertex MLP + mean/max pooling
  { GSeg s[1] = { S(sd, 7, 7) };
    run_gemm(stream, s, 1, w_lg1, b_lg1, nullptr, bufA, 64, Nv, 64, ACT_RELU); }
  { GSeg s[1] = { S(bufA, 64, 64) };
    run_gemm(stream, s, 1, w_lg2, b_lg2, nullptr, bufB, 64, Nv, 64, ACT_NONE); }
  k_pool_verts<<<dim3(Ns), dim3(64), 0, stream>>>(bufB, sd, geom, ssz, Ns, Vp);
  { GSeg s[1] = { S(ssz, 14, 14) };
    run_gemm(stream, s, 1, w_sf1, b_sf1, nullptr, bufA, 128, Ns, 128, ACT_RELU); }
  { GSeg s[1] = { S(bufA, 128, 128) };
    run_gemm(stream, s, 1, w_sf2, b_sf2, nullptr, sse, 128, Ns, 128, ACT_NONE); }

  // ==== 2. GAT on chemical graph
  { GSeg s[1] = { S(x, 6, 6) };
    run_gemm(stream, s, 1, w_atm, b_atm, nullptr, h0, 16, N, 16, ACT_NONE); }
  { GSeg s[1] = { S(eattr, 3, 3) };
    run_gemm(stream, s, 1, w_bnd, b_bnd, nullptr, ebb, 16, Eb, 16, ACT_NONE); }
  { GSeg s[1] = { S(h0, 16, 16) };
    run_gemm(stream, s, 1, w_gat, b_gat, nullptr, hg, 128, N, 128, ACT_NONE); }
  k_gat_dots<<<dim3(ceildiv(N, 256)), dim3(256), 0, stream>>>(hg, gat_as, gat_ad, asd, addt, N);
  k_fill_u32<<<dim3(ceildiv(N, 256)), dim3(256), 0, stream>>>(mx, 0x007FFFFFu, N); // enc(-inf)
  hipMemsetAsync(denom, 0, (size_t)N * 4, stream);
  hipMemsetAsync(hA, 0, (size_t)N * 128 * 4, stream);
  k_gat_logit<<<dim3(ceildiv(Eb, 256)), dim3(256), 0, stream>>>(asd, addt, bsrc, bdst, logit, mx, Eb);
  k_gat_exp<<<dim3(ceildiv(Eb, 256)), dim3(256), 0, stream>>>(logit, mx, bdst, denom, Eb);
  k_gat_agg<<<dim3(Eb), dim3(128), 0, stream>>>(logit, denom, hg, bsrc, bdst, hA, Eb);
  k_relu<<<dim3(ceildiv(N * 128, 256)), dim3(256), 0, stream>>>(hA, (long)N * 128);
  hipMemcpyAsync(cpos, pos, (size_t)N * 3 * 4, hipMemcpyDeviceToDevice, stream);

  // ==== 3. EGNN x3 on chemical graph
  float* hcur = hA; float* hnext = hB;
  for (int l = 0; l < 3; ++l) {
    k_edge_geom_bond<<<dim3(ceildiv(Eb, 256)), dim3(256), 0, stream>>>(cpos, bsrc, bdst, diffb, r2b, Eb);
    { GSeg s[4] = { S(hcur, 128, 128, bsrc), S(hcur, 128, 128, bdst), S(r2b, 1, 1), S(ebb, 16, 16) };
      run_gemm(stream, s, 4, w_me1[l], b_me1[l], nullptr, bufA, 128, Eb, 128, ACT_SILU); }
    { GSeg s[1] = { S(bufA, 128, 128) };
      run_gemm(stream, s, 1, w_me2[l], b_me2[l], nullptr, bufB, 128, Eb, 128, ACT_SILU); } // m
    { GSeg s[1] = { S(bufB, 128, 128) };
      run_gemm(stream, s, 1, w_mx1[l], b_mx1[l], nullptr, bufA, 128, Eb, 128, ACT_SILU); }
    { GSeg s[1] = { S(bufA, 128, 128) };
      run_gemm(stream, s, 1, w_mx2[l], b_mx2[l], nullptr, coef, 1, Eb, 1, ACT_NONE); }
    hipMemsetAsync(cacc, 0, (size_t)N * 3 * 4, stream);
    hipMemsetAsync(ccnt, 0, (size_t)N * 4, stream);
    k_coord_acc<<<dim3(ceildiv(Eb * 3, 256)), dim3(256), 0, stream>>>(diffb, coef, bdst, cacc, ccnt, Eb, 1);
    k_apply_coord<<<dim3(ceildiv(N * 3, 256)), dim3(256), 0, stream>>>(cpos, cacc, ccnt, N, 3);
    hipMemsetAsync(aggN, 0, (size_t)N * 128 * 4, stream);
    k_seg_sum128<<<dim3(Eb), dim3(128), 0, stream>>>(bufB, bdst, aggN, Eb);
    { GSeg s[2] = { S(hcur, 128, 128), S(aggN, 128, 128) };
      run_gemm(stream, s, 2, w_mh1[l], b_mh1[l], nullptr, u1, 128, N, 128, ACT_SILU); }
    { GSeg s[1] = { S(u1, 128, 128) };
      run_gemm(stream, s, 1, w_mh2[l], b_mh2[l], hcur, hnext, 128, N, 128, ACT_NONE); }
    float* t = hcur; hcur = hnext; hnext = t;
  }

  // ==== 4. geo MLP -> hsbuf slice 0
  { GSeg s[3] = { S(geom, 128, 128), S(sse, 128, 128), S(hcur, 128, 128, sidx) };
    run_gemm(stream, s, 3, w_ge1, b_ge1, nullptr, bufA, 128, Ns, 128, ACT_RELU); }
  { GSeg s[1] = { S(bufA, 128, 128) };
    run_gemm(stream, s, 1, w_ge2, b_ge2, nullptr, hsbuf, 640, Ns, 128, ACT_NONE); }

  // ==== 5. SurfaceEGNN x4 (2-channel coords)
  k_init_xc<<<dim3(ceildiv(Ns * 3, 256)), dim3(256), 0, stream>>>(pos, scp, sidx, xc, Ns);
  for (int l = 0; l < 4; ++l) {
    const float* hs = hsbuf + (size_t)l * 128;       // [Ns,128] view, ld 640
    float* hn = hsbuf + (size_t)(l + 1) * 128;
    k_edge_geom_surf<<<dim3(ceildiv(Es, 256)), dim3(256), 0, stream>>>(xc, ssrc, sdst, diffb, r2b, Es);
    { GSeg s[3] = { S(hs, 128, 640, ssrc), S(hs, 128, 640, sdst), S(r2b, 2, 2) };
      run_gemm(stream, s, 3, w_se1[l], b_se1[l], nullptr, bufA, 128, Es, 128, ACT_SILU); }
    { GSeg s[1] = { S(bufA, 128, 128) };
      run_gemm(stream, s, 1, w_se2[l], b_se2[l], nullptr, bufB, 128, Es, 128, ACT_SILU); } // m
    { GSeg s[1] = { S(bufB, 128, 128) };
      run_gemm(stream, s, 1, w_sx1[l], b_sx1[l], nullptr, bufA, 128, Es, 128, ACT_SILU); }
    { GSeg s[1] = { S(bufA, 128, 128) };
      run_gemm(stream, s, 1, w_sx2[l], b_sx2[l], nullptr, coef, 2, Es, 2, ACT_NONE); }
    hipMemsetAsync(cacc, 0, (size_t)Ns * 6 * 4, stream);
    hipMemsetAsync(ccnt, 0, (size_t)Ns * 4, stream);
    k_coord_acc<<<dim3(ceildiv(Es * 6, 256)), dim3(256), 0, stream>>>(diffb, coef, sdst, cacc, ccnt, Es, 2);
    k_apply_coord<<<dim3(ceildiv(Ns * 6, 256)), dim3(256), 0, stream>>>(xc, cacc, ccnt, Ns, 6);
    hipMemsetAsync(aggN, 0, (size_t)Ns * 128 * 4, stream);
    k_seg_sum128<<<dim3(Es), dim3(128), 0, stream>>>(bufB, sdst, aggN, Es);
    { GSeg s[2] = { S(hs, 128, 640), S(aggN, 128, 128) };
      run_gemm(stream, s, 2, w_sh1[l], b_sh1[l], nullptr, u1, 128, Ns, 128, ACT_SILU); }
    { GSeg s[1] = { S(u1, 128, 128) };
      run_gemm(stream, s, 1, w_sh2[l], b_sh2[l], hs, hn, 640, Ns, 128, ACT_NONE); }
  }

  // ==== 6. cutoff-attention gating + output head
  { GSeg s[1] = { S(cr, 6, 6, sidx) };
    run_gemm(stream, s, 1, w_at1, b_at1, nullptr, bufB, 128, Ns, 128, ACT_RELU); }
  { GSeg s[1] = { S(bufB, 128, 128) };
    run_gemm(stream, s, 1, w_at2, b_at2, nullptr, atts, 5, Ns, 5, ACT_SIG); }
  k_gate<<<dim3(Ns), dim3(128), 0, stream>>>(hsbuf, atts, Ns);
  { GSeg s[1] = { S(hsbuf, 640, 640) };
    run_gemm(stream, s, 1, w_ou1, b_ou1, nullptr, bufA, 640, Ns, 640, ACT_RELU); }
  { GSeg s[1] = { S(bufA, 640, 640) };
    run_gemm(stream, s, 1, w_ou2, b_ou2, nullptr, (float*)d_out, 1, Ns, 1, ACT_NONE); }
  k_angle<<<dim3(ceildiv(Ns * 3, 256)), dim3(256), 0, stream>>>(xc, pos, sidx, (float*)d_out + Ns, Ns);
}
